// GNN_Model_GIN_86045374808687
// MI455X (gfx1250) — matrix operand order repaired and checked by reading, not compiled, not run on hardware
//
#include <hip/hip_runtime.h>
#include <hip/hip_bf16.h>
#include <math.h>

typedef __attribute__((ext_vector_type(16))) __bf16 v16bf;
typedef __attribute__((ext_vector_type(8)))  float  v8f;
typedef unsigned short u16;
typedef unsigned int   u32;
typedef unsigned long long u64;
typedef int i32x2 __attribute__((vector_size(8)));

#define N_NODES 50000
#define N_EDGES 600000
#define M_PAD   50016   // ceil(N/32)*32

#if defined(__HIP_DEVICE_COMPILE__) && __has_builtin(__builtin_amdgcn_global_load_async_to_lds_b64)
#define USE_ASYNC_LDS 1
#else
#define USE_ASYNC_LDS 0
#endif

__device__ __forceinline__ u16 f2bf(float f) {
  u32 u = __float_as_uint(f);
  u = u + 0x7FFFu + ((u >> 16) & 1u);  // round-to-nearest-even
  return (u16)(u >> 16);
}

__device__ __forceinline__ float wave_sum(float v) {
#pragma unroll
  for (int off = 16; off > 0; off >>= 1) v += __shfl_xor(v, off, 32);
  return v;
}

// 8-byte global -> LDS stage (async DMA when available)
__device__ __forceinline__ void stage8(const u16* __restrict__ src, u16* dst) {
#if USE_ASYNC_LDS
  __builtin_amdgcn_global_load_async_to_lds_b64(
      (__attribute__((address_space(1))) i32x2*)src,
      (__attribute__((address_space(3))) i32x2*)dst, 0, 0);
#else
  *(u64*)dst = *(const u64*)src;
#endif
}

__device__ __forceinline__ void wait_async_le1() {
#if USE_ASYNC_LDS
#if __has_builtin(__builtin_amdgcn_s_wait_asynccnt)
  __builtin_amdgcn_s_wait_asynccnt(1);
#else
  asm volatile("s_wait_asynccnt 0x1" ::: "memory");
#endif
#endif
}
__device__ __forceinline__ void wait_async_0() {
#if USE_ASYNC_LDS
#if __has_builtin(__builtin_amdgcn_s_wait_asynccnt)
  __builtin_amdgcn_s_wait_asynccnt(0);
#else
  asm volatile("s_wait_asynccnt 0x0" ::: "memory");
#endif
#endif
}

// ---------------------------------------------------------------------------
// Fuse + cast + pad: Abf[row][k] = bf16( s*A1 + A2 ), rows/k zero-padded.
// Output is [Mpad x Kpad] bf16, 16B-vector stores.
// ---------------------------------------------------------------------------
template <bool FUSE>
__global__ __launch_bounds__(256)
void cast_pad(const float* __restrict__ A1, const float* __restrict__ A2,
              const float* __restrict__ epsPtr, u16* __restrict__ out,
              int M, int Mpad, int K, int Kpad) {
  float s = 1.0f;
  if (FUSE && epsPtr) s += epsPtr[0];
  const int cpr = Kpad >> 3;  // 8-element chunks per row
  int total = Mpad * cpr;
  for (int i = blockIdx.x * blockDim.x + threadIdx.x; i < total;
       i += gridDim.x * blockDim.x) {
    int row = i / cpr;
    int c0 = (i - row * cpr) << 3;
    u16 b[8];
    if (row < M) {
#pragma unroll
      for (int j = 0; j < 8; ++j) {
        int k = c0 + j;
        float v = 0.f;
        if (k < K) {
          size_t off = (size_t)row * K + k;
          v = A1[off];
          if (FUSE) v = s * v + A2[off];
        }
        b[j] = f2bf(v);
      }
    } else {
#pragma unroll
      for (int j = 0; j < 8; ++j) b[j] = 0;
    }
    *(uint4*)(out + (size_t)row * Kpad + c0) = *(const uint4*)b;
  }
}

// ---------------------------------------------------------------------------
// Pack W[K][HO] (f32 row-major) into WMMA B-fragment lane layout, bf16.
// idx = ((nt*KT + kt)*32 + lane)*16 + e ;  n = nt*16+(lane&15);
// k = kt*32 + (lane>=16?8:0) + (e&7) + (e>=8?16:0) ; zero-pad k>=K.
// ---------------------------------------------------------------------------
__global__ __launch_bounds__(256) void pack_w(const float* __restrict__ W,
                                              u16* __restrict__ out, int K, int HO) {
  int KT = (K + 31) >> 5;
  int total = HO * KT * 32;
  for (int i = blockIdx.x * blockDim.x + threadIdx.x; i < total;
       i += gridDim.x * blockDim.x) {
    int e = i & 15;
    int lane = (i >> 4) & 31;
    int rest = i >> 9;
    int kt = rest % KT;
    int nt = rest / KT;
    int n = nt * 16 + (lane & 15);
    int k = kt * 32 + ((lane >= 16) ? 8 : 0) + (e & 7) + ((e & 8) ? 16 : 0);
    out[i] = (k < K) ? f2bf(W[(size_t)k * HO + n]) : (u16)0;
  }
}

// ---------------------------------------------------------------------------
// Edge scatter: one wave per edge, lanes stride features (coalesced atomics).
// ---------------------------------------------------------------------------
__global__ __launch_bounds__(256) void scatter_add(const float* __restrict__ x,
                                                   const int* __restrict__ src,
                                                   const int* __restrict__ dst,
                                                   float* __restrict__ agg,
                                                   int E, int K) {
  int wid = (blockIdx.x * blockDim.x + threadIdx.x) >> 5;
  int lane = threadIdx.x & 31;
  if (wid >= E) return;
  const float* xs = x + (size_t)src[wid] * K;
  float* ad = agg + (size_t)dst[wid] * K;
  for (int k = lane; k < K; k += 32) atomicAdd(&ad[k], xs[k]);
}

// ---------------------------------------------------------------------------
// WMMA GEMM on pre-cast bf16 activations (padded: no bounds checks in loop).
// Double-buffered async global->LDS staging of 32x32 A chunks.
// Block = 32 rows x HO cols, 256 threads = 8 wave32s.
// ---------------------------------------------------------------------------
template <int HO, bool LN, bool RELU, bool RES>
__global__ __launch_bounds__(256)
void gemm_bf16_wmma(const u16* __restrict__ Abf, int Kpad,
                    const u16* __restrict__ Bp, const float* __restrict__ bias,
                    const float* __restrict__ gamma, const float* __restrict__ beta,
                    const float* __restrict__ resid, float* __restrict__ Out,
                    int M) {
  constexpr int NT  = HO / 16;  // N tiles
  constexpr int TPW = NT / 8;   // tiles per wave
  __shared__ __align__(16) u16 Asm[2][32 * 32];
  __shared__ float Csm[LN ? 32 * HO : 1];

  const int lane = threadIdx.x & 31;
  const int wave = threadIdx.x >> 5;
  const int m0 = blockIdx.x * 32;
  const int KT = Kpad >> 5;

  // staging addresses for this thread: 8 bytes (4 bf16) per thread per chunk
  const int srow = threadIdx.x >> 3;         // 0..31
  const int scol = (threadIdx.x & 7) << 2;   // 0,4,..,28 (bf16 elems)
  const u16* srcBase = Abf + (size_t)(m0 + srow) * Kpad + scol;
  u16* dst0 = &Asm[0][srow * 32 + scol];
  u16* dst1 = &Asm[1][srow * 32 + scol];

  v8f acc[TPW][2] = {};
  const int koff = (lane < 16) ? 0 : 8;  // ISA 16-bit A-fragment sub-offset
  const int arow0 = lane & 15;

  stage8(srcBase, dst0);  // prologue: chunk 0 -> buf 0
#if !USE_ASYNC_LDS
  __syncthreads();
#endif

  for (int kt = 0; kt < KT; ++kt) {
    const int cur = kt & 1;
    if (kt + 1 < KT) {  // stage next chunk into the other buffer
      stage8(srcBase + ((kt + 1) << 5), cur ? dst0 : dst1);
      wait_async_le1();   // chunk kt complete (ours); kt+1 may be in flight
    } else {
      wait_async_0();
    }
    __syncthreads();      // all waves' chunk-kt stores visible

    union Frag { v16bf v; uint4 q[2]; };
    Frag a[2];
#pragma unroll
    for (int mt = 0; mt < 2; ++mt) {
      const u16* rp = &Asm[cur][(mt * 16 + arow0) * 32];
      a[mt].q[0] = *(const uint4*)(rp + koff);       // K = koff..koff+7
      a[mt].q[1] = *(const uint4*)(rp + koff + 16);  // K = koff+16..koff+23
    }
#pragma unroll
    for (int t = 0; t < TPW; ++t) {
      int nt = wave + t * 8;
      const uint4* bp =
          (const uint4*)(Bp + (((size_t)nt * KT + kt) * 32 + lane) * 16);
      Frag b;
      b.q[0] = bp[0];
      b.q[1] = bp[1];
#pragma unroll
      for (int mt = 0; mt < 2; ++mt)
        acc[t][mt] = __builtin_amdgcn_wmma_f32_16x16x32_bf16(
            false, a[mt].v, false, b.v, (short)0, acc[t][mt], false, false);
    }
    __syncthreads();  // protect buffer `cur` before it is re-staged at kt+2
  }

  if constexpr (LN) {
#pragma unroll
    for (int t = 0; t < TPW; ++t) {
      int nt = wave + t * 8;
      int n = nt * 16 + (lane & 15);
#pragma unroll
      for (int mt = 0; mt < 2; ++mt) {
        int rbase = mt * 16 + ((lane < 16) ? 0 : 8);
#pragma unroll
        for (int r = 0; r < 8; ++r)
          Csm[(rbase + r) * HO + n] = acc[t][mt][r] + bias[n];
      }
    }
    __syncthreads();
#pragma unroll
    for (int rr = 0; rr < 4; ++rr) {
      int row = wave * 4 + rr;
      int gm = m0 + row;
      const float* cr = &Csm[row * HO];
      float sum = 0.f;
      for (int c = lane; c < HO; c += 32) sum += cr[c];
      float mean = wave_sum(sum) * (1.0f / HO);
      float vs = 0.f;
      for (int c = lane; c < HO; c += 32) { float d = cr[c] - mean; vs += d * d; }
      float rstd = rsqrtf(wave_sum(vs) * (1.0f / HO) + 1e-5f);
      if (gm < M) {
        for (int c = lane; c < HO; c += 32) {
          float v = (cr[c] - mean) * rstd * gamma[c] + beta[c];
          v = fmaxf(v, 0.f);
          Out[(size_t)gm * HO + c] = v;
        }
      }
    }
  } else {
#pragma unroll
    for (int t = 0; t < TPW; ++t) {
      int nt = wave + t * 8;
      int n = nt * 16 + (lane & 15);
#pragma unroll
      for (int mt = 0; mt < 2; ++mt) {
        int mb = m0 + mt * 16 + ((lane < 16) ? 0 : 8);
#pragma unroll
        for (int r = 0; r < 8; ++r) {
          int m = mb + r;
          if (m < M) {
            float v = acc[t][mt][r] + bias[n];
            if constexpr (RELU) v = fmaxf(v, 0.f);
            if constexpr (RES) v += resid[(size_t)m * HO + n];
            Out[(size_t)m * HO + n] = v;
          }
        }
      }
    }
  }
}

// ---------------------------------------------------------------------------
// Attention gate: att = sigmoid(h4.Wa + b); hw = h4 * att (wave per row)
// ---------------------------------------------------------------------------
__global__ __launch_bounds__(256) void att_kernel(const float* __restrict__ h4,
                                                  const float* __restrict__ Wa,
                                                  const float* __restrict__ ba,
                                                  float* __restrict__ hw, int Nn) {
  int wid = (blockIdx.x * blockDim.x + threadIdx.x) >> 5;
  int lane = threadIdx.x & 31;
  if (wid >= Nn) return;
  const float* hr = h4 + (size_t)wid * 128;
  float d = 0.f;
  float hv[4];
#pragma unroll
  for (int i = 0; i < 4; ++i) {
    hv[i] = hr[lane + 32 * i];
    d += hv[i] * Wa[lane + 32 * i];
  }
  d = wave_sum(d);
  float a = 1.f / (1.f + expf(-(d + ba[0])));
#pragma unroll
  for (int i = 0; i < 4; ++i) hw[(size_t)wid * 128 + lane + 32 * i] = hv[i] * a;
}

// ---------------------------------------------------------------------------
// 128 -> 1 head. mode 0: tanh(.)*pi   mode 1: 0.9 + 0.2*sigmoid(.)
// ---------------------------------------------------------------------------
__global__ __launch_bounds__(256) void gemv_head(const float* __restrict__ X,
                                                 const float* __restrict__ W,
                                                 const float* __restrict__ b,
                                                 float* __restrict__ out, int Nn,
                                                 int mode) {
  int wid = (blockIdx.x * blockDim.x + threadIdx.x) >> 5;
  int lane = threadIdx.x & 31;
  if (wid >= Nn) return;
  const float* xr = X + (size_t)wid * 128;
  float d = 0.f;
#pragma unroll
  for (int i = 0; i < 4; ++i) d += xr[lane + 32 * i] * W[lane + 32 * i];
  d = wave_sum(d);
  if (lane == 0) {
    float v = d + b[0];
    out[wid] = (mode == 0) ? tanhf(v) * 3.14159265358979323846f
                           : 0.9f + 0.2f / (1.f + expf(-v));
  }
}

// ---------------------------------------------------------------------------
__global__ __launch_bounds__(256) void coords_kernel(const float* __restrict__ ang,
                                                     const float* __restrict__ rad,
                                                     float* __restrict__ out,
                                                     float* __restrict__ sums,
                                                     int Nn) {
  __shared__ float s0[8], s1[8];
  int i = blockIdx.x * 256 + threadIdx.x;
  int lane = threadIdx.x & 31, wave = threadIdx.x >> 5;
  float c0 = 0.f, c1 = 0.f;
  if (i < Nn) {
    float a = ang[i], r = rad[i];
    c0 = r * cosf(a);
    c1 = r * sinf(a);
    out[2 * i] = c0;
    out[2 * i + 1] = c1;
  }
  c0 = wave_sum(c0);
  c1 = wave_sum(c1);
  if (lane == 0) { s0[wave] = c0; s1[wave] = c1; }
  __syncthreads();
  if (threadIdx.x == 0) {
    float t0 = 0.f, t1 = 0.f;
    for (int w = 0; w < 8; ++w) { t0 += s0[w]; t1 += s1[w]; }
    atomicAdd(&sums[0], t0);
    atomicAdd(&sums[1], t1);
  }
}

__global__ __launch_bounds__(256) void center_kernel(float* __restrict__ out,
                                                     const float* __restrict__ sums,
                                                     int Nn) {
  int i = blockIdx.x * 256 + threadIdx.x;
  if (i < Nn) {
    out[2 * i]     -= sums[0] / (float)Nn;
    out[2 * i + 1] -= sums[1] / (float)Nn;
  }
}

// ---------------------------------------------------------------------------
extern "C" void kernel_launch(void* const* d_in, const int* in_sizes, int n_in,
                              void* d_out, int out_size, void* d_ws, size_t ws_size,
                              hipStream_t stream) {
  (void)in_sizes; (void)n_in; (void)out_size; (void)ws_size;
  const float* x = (const float*)d_in[0];
  const int* ei = (const int*)d_in[1];
  const int* src = ei;
  const int* dst = ei + N_EDGES;

  struct Conv { const float *eps, *W1, *b1, *g1, *be1, *W2, *b2; };
  Conv cv[4];
  for (int i = 0; i < 4; ++i) {
    int b = 2 + i * 7;
    cv[i] = { (const float*)d_in[b],     (const float*)d_in[b + 1],
              (const float*)d_in[b + 2], (const float*)d_in[b + 3],
              (const float*)d_in[b + 4], (const float*)d_in[b + 5],
              (const float*)d_in[b + 6] };
  }
  const float* attW = (const float*)d_in[30];
  const float* attB = (const float*)d_in[31];
  const float* aW1 = (const float*)d_in[32]; const float* aB1  = (const float*)d_in[33];
  const float* aG1 = (const float*)d_in[34]; const float* aBe1 = (const float*)d_in[35];
  const float* aW2 = (const float*)d_in[36]; const float* aB2  = (const float*)d_in[37];
  const float* aG2 = (const float*)d_in[38]; const float* aBe2 = (const float*)d_in[39];
  const float* aW3 = (const float*)d_in[40]; const float* aB3  = (const float*)d_in[41];
  const float* rW1 = (const float*)d_in[42]; const float* rB1  = (const float*)d_in[43];
  const float* rG1 = (const float*)d_in[44]; const float* rBe1 = (const float*)d_in[45];
  const float* rW2 = (const float*)d_in[46]; const float* rB2  = (const float*)d_in[47];

  // ---- workspace layout (float units) ----
  float* ws = (float*)d_ws;
  size_t o = 0;
  float* agg  = ws + o; o += (size_t)N_NODES * 257;
  float* tbuf = ws + o; o += (size_t)N_NODES * 256;
  float* hA   = ws + o; o += (size_t)N_NODES * 128;
  float* hB   = ws + o; o += (size_t)N_NODES * 128;
  float* hw   = ws + o; o += (size_t)N_NODES * 128;
  float* ang  = ws + o; o += N_NODES;
  float* rad  = ws + o; o += N_NODES;
  float* sums = ws + o; o += 8;
  u16* Abf = (u16*)(ws + o); o += (size_t)M_PAD * 288 / 2;  // bf16 activations
  u16* pk = (u16*)(ws + o);
  size_t po = 0;
  u16* pk_w1[4]; u16* pk_w2[4];
  pk_w1[0] = pk + po; po += (size_t)128 * 9 * 32;  // K=257 -> KT=9
  pk_w2[0] = pk + po; po += (size_t)128 * 4 * 32;
  for (int L = 1; L < 4; ++L) {
    pk_w1[L] = pk + po; po += (size_t)128 * 4 * 32;
    pk_w2[L] = pk + po; po += (size_t)128 * 4 * 32;
  }
  u16* pk_aW1 = pk + po; po += (size_t)256 * 4 * 32;
  u16* pk_aW2 = pk + po; po += (size_t)128 * 8 * 32;
  u16* pk_rW1 = pk + po; po += (size_t)128 * 4 * 32;

  auto pack = [&](const float* W, u16* outp, int K, int HO) {
    int KT = (K + 31) / 32;
    int total = HO * KT * 32;
    pack_w<<<(total + 255) / 256, 256, 0, stream>>>(W, outp, K, HO);
  };
  pack(cv[0].W1, pk_w1[0], 257, 128);
  pack(cv[0].W2, pk_w2[0], 128, 128);
  for (int L = 1; L < 4; ++L) {
    pack(cv[L].W1, pk_w1[L], 128, 128);
    pack(cv[L].W2, pk_w2[L], 128, 128);
  }
  pack(aW1, pk_aW1, 128, 256);
  pack(aW2, pk_aW2, 256, 128);
  pack(rW1, pk_rW1, 128, 128);

  const int GB = M_PAD / 32;          // 1563 row-blocks
  const int EB = (N_EDGES + 7) / 8;   // wave per edge
  const int WB = (N_NODES + 7) / 8;   // wave per node
  auto castGrid = [](int Kpad) { return (int)(((size_t)M_PAD * (Kpad / 8) + 255) / 256); };

  // ---- conv1 (K = 257, Kpad = 288) ----
  (void)hipMemsetAsync(agg, 0, (size_t)N_NODES * 257 * sizeof(float), stream);
  scatter_add<<<EB, 256, 0, stream>>>(x, src, dst, agg, N_EDGES, 257);
  cast_pad<true><<<castGrid(288), 256, 0, stream>>>(x, agg, cv[0].eps, Abf,
                                                    N_NODES, M_PAD, 257, 288);
  gemm_bf16_wmma<128, true, true, false><<<GB, 256, 0, stream>>>(
      Abf, 288, pk_w1[0], cv[0].b1, cv[0].g1, cv[0].be1, nullptr, tbuf, N_NODES);
  cast_pad<false><<<castGrid(128), 256, 0, stream>>>(tbuf, nullptr, nullptr, Abf,
                                                     N_NODES, M_PAD, 128, 128);
  gemm_bf16_wmma<128, false, true, false><<<GB, 256, 0, stream>>>(
      Abf, 128, pk_w2[0], cv[0].b2, nullptr, nullptr, nullptr, hA, N_NODES);

  // ---- conv2..conv4 (K = 128, relu + residual) ----
  float* hcur = hA;
  float* hnext = hB;
  for (int L = 1; L < 4; ++L) {
    (void)hipMemsetAsync(agg, 0, (size_t)N_NODES * 128 * sizeof(float), stream);
    scatter_add<<<EB, 256, 0, stream>>>(hcur, src, dst, agg, N_EDGES, 128);
    cast_pad<true><<<castGrid(128), 256, 0, stream>>>(hcur, agg, cv[L].eps, Abf,
                                                      N_NODES, M_PAD, 128, 128);
    gemm_bf16_wmma<128, true, true, false><<<GB, 256, 0, stream>>>(
        Abf, 128, pk_w1[L], cv[L].b1, cv[L].g1, cv[L].be1, nullptr, tbuf, N_NODES);
    cast_pad<false><<<castGrid(128), 256, 0, stream>>>(tbuf, nullptr, nullptr, Abf,
                                                       N_NODES, M_PAD, 128, 128);
    gemm_bf16_wmma<128, false, true, true><<<GB, 256, 0, stream>>>(
        Abf, 128, pk_w2[L], cv[L].b2, nullptr, nullptr, hcur, hnext, N_NODES);
    float* tmp = hcur; hcur = hnext; hnext = tmp;
  }
  // hcur == h4, hnext free

  // ---- attention gate ----
  att_kernel<<<WB, 256, 0, stream>>>(hcur, attW, attB, hw, N_NODES);

  // ---- angle head ----
  cast_pad<false><<<castGrid(128), 256, 0, stream>>>(hw, nullptr, nullptr, Abf,
                                                     N_NODES, M_PAD, 128, 128);
  gemm_bf16_wmma<256, true, true, false><<<GB, 256, 0, stream>>>(
      Abf, 128, pk_aW1, aB1, aG1, aBe1, nullptr, tbuf, N_NODES);
  cast_pad<false><<<castGrid(256), 256, 0, stream>>>(tbuf, nullptr, nullptr, Abf,
                                                     N_NODES, M_PAD, 256, 256);
  gemm_bf16_wmma<128, true, true, false><<<GB, 256, 0, stream>>>(
      Abf, 256, pk_aW2, aB2, aG2, aBe2, nullptr, hnext, N_NODES);
  gemv_head<<<WB, 256, 0, stream>>>(hnext, aW3, aB3, ang, N_NODES, 0);

  // ---- radius head ----
  cast_pad<false><<<castGrid(128), 256, 0, stream>>>(hw, nullptr, nullptr, Abf,
                                                     N_NODES, M_PAD, 128, 128);
  gemm_bf16_wmma<128, true, true, false><<<GB, 256, 0, stream>>>(
      Abf, 128, pk_rW1, rB1, rG1, rBe1, nullptr, tbuf, N_NODES);
  gemv_head<<<WB, 256, 0, stream>>>(tbuf, rW2, rB2, rad, N_NODES, 1);

  // ---- coords + mean-center ----
  (void)hipMemsetAsync(sums, 0, 8 * sizeof(float), stream);
  coords_kernel<<<(N_NODES + 255) / 256, 256, 0, stream>>>(ang, rad, (float*)d_out,
                                                           sums, N_NODES);
  center_kernel<<<(N_NODES + 255) / 256, 256, 0, stream>>>((float*)d_out, sums,
                                                           N_NODES);
}
